// MoeConv_16501264351576
// MI455X (gfx1250) — matrix operand order repaired and checked by reading, not compiled, not run on hardware
//
#include <hip/hip_runtime.h>
#include <math.h>

typedef __attribute__((ext_vector_type(16))) _Float16 v16h;
typedef __attribute__((ext_vector_type(8)))  _Float16 v8h;
typedef __attribute__((ext_vector_type(8)))  float    v8f;

#define CIN   128
#define COUT  128
#define HH    96
#define WW    96
#define NE    8
#define KTOT  1152      // 9 taps * 128 ci ; k = tap*128 + ci
#define CIP   136       // Xs ci stride (halfs): 128 + 8 pad (272B, 16B-aligned)

// ---------------- kernel 0: pack expert+shared weights to f16 -----------------
// Apack[(s*128+co)][tap*128+ci], s=0..7 experts, s=8 shared.  1152x1152 halfs.
__global__ __launch_bounds__(256) void moe_pack_w_k(
    const float* __restrict__ expert_w, const float* __restrict__ shared_w,
    _Float16* __restrict__ Apack)
{
  int idx = blockIdx.x * 256 + threadIdx.x;       // exactly 1152*1152 threads
  int m   = idx / KTOT;
  int k   = idx - m * KTOT;
  int tap = k >> 7;
  int ci  = k & 127;
  int s   = m >> 7;
  int co  = m & 127;
  float v;
  if (s < NE) v = expert_w[((s * COUT + co) * CIN + ci) * 9 + tap];
  else        v = shared_w[(co * CIN + ci) * 9 + tap];
  Apack[m * KTOT + k] = (_Float16)v;
}

// ---------------- kernel 1: gate conv + sigmoid ------------------------------
__global__ __launch_bounds__(256) void moe_gate_k(
    const float* __restrict__ x, const float* __restrict__ gate_w,
    float* __restrict__ gsc)
{
  __shared__ float gwl[CIN * 9 * NE];   // [(ci*9+tap)*8 + e], 36 KB
  int tid = threadIdx.x;
  for (int i = tid; i < NE * CIN * 9; i += 256) {
    int e = i / (CIN * 9);
    int r = i - e * (CIN * 9);
    gwl[r * NE + e] = gate_w[i];
  }
  __syncthreads();

  int pix = blockIdx.x * 256 + tid;     // exactly 4*9216 threads
  int b   = pix / (HH * WW);
  int hw  = pix - b * (HH * WW);
  int h   = hw / WW;
  int w   = hw - h * WW;

  float acc[NE];
#pragma unroll
  for (int e = 0; e < NE; ++e) acc[e] = 0.f;

  for (int ci = 0; ci < CIN; ++ci) {
    const float* xp = x + ((size_t)(b * CIN + ci) * HH) * WW;
#pragma unroll
    for (int t = 0; t < 9; ++t) {
      int h2 = h + (t / 3) - 1;
      int w2 = w + (t % 3) - 1;
      if (h2 >= 0 && h2 < HH && w2 >= 0 && w2 < WW) {
        float xv = xp[h2 * WW + w2];
        const float* gp = &gwl[(ci * 9 + t) * NE];
#pragma unroll
        for (int e = 0; e < NE; ++e) acc[e] = fmaf(xv, gp[e], acc[e]);
      }
    }
  }
#pragma unroll
  for (int e = 0; e < NE; ++e)
    gsc[(size_t)(b * NE + e) * (HH * WW) + hw] = 1.f / (1.f + expf(-acc[e]));
}

// ---------------- kernel 2: top-2 routing + row activity flags ---------------
__global__ __launch_bounds__(96) void moe_route_k(
    const float* __restrict__ gsc, const float* __restrict__ gate_bias,
    float* __restrict__ score_full, int* __restrict__ rowflag)
{
  __shared__ int fl[NE];
  int row = blockIdx.x;                 // b*96 + h, 384 rows
  int b   = row / HH;
  int h   = row - b * HH;
  int tid = threadIdx.x;                // w: 0..95
  int hw  = h * WW + tid;

  float sc[NE], bi[NE];
#pragma unroll
  for (int e = 0; e < NE; ++e) {
    sc[e] = gsc[(size_t)(b * NE + e) * (HH * WW) + hw];
    bi[e] = sc[e] + gate_bias[e];
  }
  int i0 = 0; float m0b = bi[0]; float m0s = sc[0];
#pragma unroll
  for (int e = 1; e < NE; ++e)
    if (bi[e] > m0b) { m0b = bi[e]; m0s = sc[e]; i0 = e; }
  int i1 = -1; float m1b = -3.4e38f; float m1s = 0.f;
#pragma unroll
  for (int e = 0; e < NE; ++e)
    if (e != i0 && bi[e] > m1b) { m1b = bi[e]; m1s = sc[e]; i1 = e; }

  float mx  = fmaxf(m0s, m1s);
  float e0  = expf(m0s - mx), e1 = expf(m1s - mx);
  float inv = 1.f / (e0 + e1);
  float w0  = e0 * inv, w1 = e1 * inv;   // ROUTE_SCALE = 1.0

  if (tid < NE) fl[tid] = 0;
  __syncthreads();
  fl[i0] = 1; fl[i1] = 1;                // benign write race (all write 1)
  __syncthreads();

#pragma unroll
  for (int e = 0; e < NE; ++e)
    score_full[(size_t)(b * NE + e) * (HH * WW) + hw] =
        (e == i0) ? w0 : ((e == i1) ? w1 : 0.f);
  if (tid < NE) rowflag[row * NE + tid] = fl[tid];
}

// ---------------- kernel 3: WMMA implicit-GEMM + fused MoE epilogue ----------
// grid (2, 384): x = co tile (64 channels), y = b*96 + h (one image row: N=96)
// 128 threads = 4 waves: wave_m 0..1 (32 M rows each -> 2 WMMA M-subtiles),
// wave_n 0..1 (48 px -> 3 WMMA N-tiles). Each B fragment feeds 2 WMMAs
// (1 ds_load_b128 per WMMA); A fragments come straight from L2-resident
// packed weights. x slice staged ONCE per block; no barriers in hot loop.
__global__ __launch_bounds__(128) void moe_main_wmma_k(
    const float* __restrict__ x, const _Float16* __restrict__ Apack,
    const float* __restrict__ score_full, const int* __restrict__ rowflag,
    const float* __restrict__ expert_b, const float* __restrict__ shared_b,
    float* __restrict__ out)
{
  __shared__ __align__(16) _Float16 Xs[3 * 98 * CIP];   // 79,968 B

  const int tid    = threadIdx.x;
  const int lane   = tid & 31;
  const int wid    = tid >> 5;
  const int wave_m = wid >> 1;          // 0..1 -> 32 M rows
  const int wave_n = wid & 1;           // 0..1 -> 48 pixels
  const int lhalf  = lane & 15;
  const int hsel   = lane >> 4;         // 0/1

  const int co_base = blockIdx.x * 64;
  const int bh = blockIdx.y;
  const int b  = bh / HH;
  const int h  = bh - b * HH;

  // ---- one-time staging: x[b, :, h-1..h+1, :] -> Xs[r3][col+1][ci] (f16) ----
  for (int p = tid; p < 3 * 98; p += 128) {
    int r3   = p / 98;                  // 0..2 -> image row h + r3 - 1
    int cs   = p - r3 * 98;             // 0..97 -> source col cs - 1
    int wsrc = cs - 1;
    int h2   = h + r3 - 1;
    _Float16* dst = Xs + p * CIP;
    if (wsrc >= 0 && wsrc < WW && h2 >= 0 && h2 < HH) {
      const float* src = x + ((size_t)b * CIN * HH + h2) * WW + wsrc;
#pragma unroll 4
      for (int ci = 0; ci < CIN; ci += 2) {
        union { _Float16 hx[2]; unsigned int u; } pk;
        pk.hx[0] = (_Float16)src[(size_t)ci * (HH * WW)];
        pk.hx[1] = (_Float16)src[(size_t)(ci + 1) * (HH * WW)];
        *(unsigned int*)(dst + ci) = pk.u;
      }
    } else {
#pragma unroll 8
      for (int ci = 0; ci < CIN; ci += 2)
        *(unsigned int*)(dst + ci) = 0u;
    }
  }
  __syncthreads();

  const v8f zero = {0.f, 0.f, 0.f, 0.f, 0.f, 0.f, 0.f, 0.f};
  v8f fin0[3], fin1[3];
#pragma unroll
  for (int q = 0; q < 3; ++q) { fin0[q] = zero; fin1[q] = zero; }

  const int amrow = co_base + wave_m * 32 + lhalf;  // first M-subtile row
  const int bn0   = wave_n * 48 + lhalf;            // base pixel (per lane)

  for (int s = 0; s < NE + 1; ++s) {    // 8 experts + shared(=8)
    if (s < NE && rowflag[bh * NE + s] == 0) continue;   // uniform, no barriers

    v8f acc0[3], acc1[3];
#pragma unroll
    for (int q = 0; q < 3; ++q) { acc0[q] = zero; acc1[q] = zero; }

    // A fragment bases: lanes 0-15 -> k chunks {0-7,16-23}; 16-31 -> {8-15,24-31}
    const _Float16* arow0 =
        Apack + (size_t)(s * COUT + amrow) * KTOT + hsel * 8;
    const _Float16* arow1 = arow0 + (size_t)16 * KTOT;   // second M-subtile

    for (int tap = 0; tap < 9; ++tap) {
      int dh = tap / 3;                 // 0..2
      int dw = tap % 3;                 // 0..2 ; Xs col index cs = n + dw
      const _Float16* brow = Xs + (size_t)(dh * 98 + bn0 + dw) * CIP;
#pragma unroll
      for (int cq = 0; cq < 4; ++cq) {
        const int ci0 = cq * 32;
        const _Float16* ap0 = arow0 + tap * 128 + ci0;
        const _Float16* ap1 = arow1 + tap * 128 + ci0;
        v8h a0l = *(const v8h*)(ap0);
        v8h a0h = *(const v8h*)(ap0 + 16);
        v8h a1l = *(const v8h*)(ap1);
        v8h a1h = *(const v8h*)(ap1 + 16);
        v16h av0 = __builtin_shufflevector(a0l, a0h, 0, 1, 2, 3, 4, 5, 6, 7,
                                           8, 9, 10, 11, 12, 13, 14, 15);
        v16h av1 = __builtin_shufflevector(a1l, a1h, 0, 1, 2, 3, 4, 5, 6, 7,
                                           8, 9, 10, 11, 12, 13, 14, 15);
#pragma unroll
        for (int q = 0; q < 3; ++q) {
          // B fragment: lane n = bn0 + q*16; k chunks {hsel*8, 16+hsel*8} (matches A)
          const _Float16* bp = brow + (size_t)q * 16 * CIP + ci0 + hsel * 8;
          v8h blo = *(const v8h*)(bp);
          v8h bhi = *(const v8h*)(bp + 16);
          v16h bv = __builtin_shufflevector(blo, bhi, 0, 1, 2, 3, 4, 5, 6, 7,
                                            8, 9, 10, 11, 12, 13, 14, 15);
          acc0[q] = __builtin_amdgcn_wmma_f32_16x16x32_f16(
              false, av0, false, bv, (short)0, acc0[q], false, false);
          acc1[q] = __builtin_amdgcn_wmma_f32_16x16x32_f16(
              false, av1, false, bv, (short)0, acc1[q], false, false);
        }
      }
    }

    // ---- fused epilogue: fin += w_route(pixel) * (conv + bias) ----
    const int cb0 = co_base + wave_m * 32 + hsel * 8;    // C/D: M = r + 8*hsel
    const int cb1 = cb0 + 16;
    float wq[3];
#pragma unroll
    for (int q = 0; q < 3; ++q) {
      int n = bn0 + q * 16;
      wq[q] = (s == NE) ? 1.f
                        : score_full[(size_t)(b * NE + s) * (HH * WW) + h * WW + n];
    }
#pragma unroll
    for (int r = 0; r < 8; ++r) {
      float b0 = (s == NE) ? shared_b[cb0 + r] : expert_b[s * COUT + cb0 + r];
      float b1 = (s == NE) ? shared_b[cb1 + r] : expert_b[s * COUT + cb1 + r];
#pragma unroll
      for (int q = 0; q < 3; ++q) {
        fin0[q][r] += wq[q] * (acc0[q][r] + b0);
        fin1[q][r] += wq[q] * (acc1[q][r] + b1);
      }
    }
  }

  // ---- store final (B, Cout, H, W) ----
#pragma unroll
  for (int q = 0; q < 3; ++q) {
    int w = bn0 + q * 16;
#pragma unroll
    for (int r = 0; r < 8; ++r) {
      int co0 = co_base + wave_m * 32 + hsel * 8 + r;
      out[((size_t)(b * COUT + co0) * HH + h) * WW + w]      = fin0[q][r];
      out[((size_t)(b * COUT + co0 + 16) * HH + h) * WW + w] = fin1[q][r];
    }
  }
}

// ---------------- launch ------------------------------------------------------
extern "C" void kernel_launch(void* const* d_in, const int* in_sizes, int n_in,
                              void* d_out, int out_size, void* d_ws, size_t ws_size,
                              hipStream_t stream) {
  (void)in_sizes; (void)n_in; (void)out_size; (void)ws_size;
  const float* x         = (const float*)d_in[0];
  const float* gate_w    = (const float*)d_in[1];
  const float* gate_bias = (const float*)d_in[2];
  const float* expert_w  = (const float*)d_in[3];
  const float* expert_b  = (const float*)d_in[4];
  const float* shared_w  = (const float*)d_in[5];
  const float* shared_b  = (const float*)d_in[6];
  float* out = (float*)d_out;

  char* ws = (char*)d_ws;
  _Float16* Apack      = (_Float16*)(ws);                 // 1152*1152*2 = 2,654,208 B
  float*    score_full = (float*)(ws + 2654208);          // 4*8*9216*4  = 1,179,648 B
  float*    gsc        = (float*)(ws + 3833856);          // 4*8*9216*4  = 1,179,648 B
  int*      rowflag    = (int*)(ws + 5013504);            // 384*8*4     =    12,288 B

  moe_pack_w_k<<<(1152 * 1152) / 256, 256, 0, stream>>>(expert_w, shared_w, Apack);
  moe_gate_k<<<(4 * HH * WW) / 256, 256, 0, stream>>>(x, gate_w, gsc);
  moe_route_k<<<4 * HH, 96, 0, stream>>>(gsc, gate_bias, score_full, rowflag);
  moe_main_wmma_k<<<dim3(2, 4 * HH), 128, 0, stream>>>(
      x, Apack, score_full, rowflag, expert_b, shared_b, out);
}
